// update_13082470383779
// MI455X (gfx1250) — matrix operand order repaired and checked by reading, not compiled, not run on hardware
//
#include <hip/hip_runtime.h>
#include <math.h>
#include <stdint.h>

// ---------------- configuration ----------------
constexpr int kD        = 128;            // feature dim
constexpr int kTile     = 16;             // samples per WMMA tile
constexpr int kTilesPB  = 32;             // tiles per block
constexpr int kSamplesPB = kTile * kTilesPB;  // 512 samples per block
constexpr int kThreads  = 256;            // 8 wave32

typedef __bf16 bf16_t;
typedef bf16_t v16bf __attribute__((ext_vector_type(16)));
typedef bf16_t v8bf  __attribute__((ext_vector_type(8)));
typedef bf16_t v4bf  __attribute__((ext_vector_type(4)));
typedef float  v8f   __attribute__((ext_vector_type(8)));
typedef float  v4f   __attribute__((ext_vector_type(4)));
typedef unsigned int v4u __attribute__((ext_vector_type(4)));
typedef int    v8i   __attribute__((ext_vector_type(8)));
typedef int    v4i   __attribute__((ext_vector_type(4)));

// LDS layout (offsets in bf16 elements unless noted)
constexpr int OFF_VW  = 0;                      // [128][128] bf16
constexpr int OFF_UW  = OFF_VW  + kD * kD;      // [128][128]
constexpr int OFF_A1W = OFF_UW  + kD * kD;      // [128][256]
constexpr int OFF_A2W = OFF_A1W + kD * 2*kD;    // [384][128]
constexpr int OFF_V1  = OFF_A2W + 3*kD * kD;    // [48][128]  v1 bf16
constexpr int OFF_SS  = OFF_V1  + 48 * kD;      // [16][256]  s_stack = [vnorm | s]
constexpr int OFF_H   = OFF_SS  + kTile * 2*kD; // [16][128]  h bf16
// f32 staging (double buffered); TDM target. bf16 X operand reuses low half in place.
constexpr int STAGE_BYTE   = (OFF_H + kTile * kD) * 2;   // 253,952 (16B aligned)
constexpr int STAGE_FLOATS = 48*kD + kTile*kD;           // 8192 floats = 32 KB / buffer
constexpr int LDS_BYTES    = STAGE_BYTE + 2 * STAGE_FLOATS * 4;  // 319,488 <= 320 KB

// A-operand fragment: 16x32 bf16 tile, lane = row M, K-groups per ISA 7.12.2
__device__ __forceinline__ v16bf load_a_frag(const bf16_t* base, int row, int stride,
                                             int k0, int laneHi) {
    const bf16_t* p = base + row * stride + k0 + (laneHi ? 8 : 0);
    v8bf lo = *(const v8bf*)(p);
    v8bf hi = *(const v8bf*)(p + 16);
    return __builtin_shufflevector(lo, hi, 0,1,2,3,4,5,6,7,8,9,10,11,12,13,14,15);
}

// B-operand fragment: 32x16 bf16 tile; lane = output column = W row
__device__ __forceinline__ v16bf load_b_frag(const bf16_t* base, int wrow, int stride,
                                             int k0, int laneHi) {
    const bf16_t* p = base + wrow * stride + k0 + (laneHi ? 8 : 0);
    v8bf lo = *(const v8bf*)(p);
    v8bf hi = *(const v8bf*)(p + 16);
    return __builtin_shufflevector(lo, hi, 0,1,2,3,4,5,6,7,8,9,10,11,12,13,14,15);
}

#define WMMA_BF16(A, B, C) \
    __builtin_amdgcn_wmma_f32_16x16x32_bf16(false, (A), false, (B), (short)0, (C), false, false)

// Issue one TDM 2D tile load: rows x 128 f32, row stride strideElems, zero-fill
// rows beyond tensorRows. D# bitfields per CDNA5 ISA 8.3-8.5.
__device__ __forceinline__ void tdm_load_tile(unsigned ldsByte, const float* gsrc,
                                              int tensorRows, int strideElems) {
    unsigned long long ga = (unsigned long long)(uintptr_t)gsrc;
    v4u g0;
    g0[0] = 1u;                                          // count=1, user mode, no gather
    g0[1] = ldsByte;                                     // lds_addr (bytes)
    g0[2] = (unsigned)(ga & 0xffffffffu);                // global_addr[31:0]
    g0[3] = (unsigned)((ga >> 32) & 0x01ffffffu) | (2u << 30);  // addr[56:32] | type=2
    v8i g1;
    g1[0] = (int)(2u << 16);                             // wg_mask=0, data_size=4B
    g1[1] = (int)(128u << 16);                           // tensor_dim0 = 128 (low16)
    g1[2] = (int)((unsigned)tensorRows << 16);           // tensor_dim1 = valid rows
    g1[3] = (int)(128u << 16);                           // tile_dim0 = 128
    g1[4] = 16;                                          // tile_dim1 = 16, tile_dim2 = 0
    g1[5] = strideElems;                                 // tensor_dim0_stride (low 32)
    g1[6] = 0;
    g1[7] = 0;
    v4i gz4 = {0, 0, 0, 0};
    v8i gz8 = {0, 0, 0, 0, 0, 0, 0, 0};
    __builtin_amdgcn_tensor_load_to_lds(g0, g1, gz4, gz4, gz8, 0);
}

// Prefetch one 16-sample tile (3 v-channel tiles + s tile) into staging buffer.
__device__ __forceinline__ void prefetch_tile(const float* v, const float* s,
                                              long long Sb, int nTotal, int bufByte) {
    long long rem = (long long)nTotal - Sb;
    if (rem <= 0) return;
    int rows = rem < kTile ? (int)rem : kTile;
    for (int c = 0; c < 3; ++c)
        tdm_load_tile((unsigned)(bufByte + c * kTile * kD * 4),
                      v + (Sb * 3 + c) * kD, rows, 3 * kD);
    tdm_load_tile((unsigned)(bufByte + 48 * kD * 4), s + Sb * kD, rows, kD);
}

__device__ __forceinline__ void stage_w(bf16_t* dst, const float* src, int count, int tid) {
    for (int i = tid * 4; i < count; i += kThreads * 4) {
        v4f f = *(const v4f*)(src + i);
        v4bf o;
        o[0] = (bf16_t)f[0]; o[1] = (bf16_t)f[1]; o[2] = (bf16_t)f[2]; o[3] = (bf16_t)f[3];
        *(v4bf*)(dst + i) = o;
    }
}

extern "C" __global__ __launch_bounds__(kThreads)
void painn_fused_kernel(const float* __restrict__ s,   const float* __restrict__ v,
                        const float* __restrict__ Vw,  const float* __restrict__ Vb,
                        const float* __restrict__ Uw,  const float* __restrict__ Ub,
                        const float* __restrict__ a1w, const float* __restrict__ a1b,
                        const float* __restrict__ a2w, const float* __restrict__ a2b,
                        float* __restrict__ out, int nTotal)
{
    extern __shared__ __align__(16) bf16_t lds[];

    const int tid    = threadIdx.x;
    const int wave   = tid >> 5;
    const int lane   = tid & 31;
    const int laneN  = lane & 15;
    const int laneHi = lane >> 4;

    const long long base0 = (long long)blockIdx.x * kSamplesPB;

    // kick off TDM for tile 0 immediately; it overlaps with weight staging
    if (wave == 0) prefetch_tile(v, s, base0, nTotal, STAGE_BYTE);

    // ---- stage all weights to LDS as bf16 (b128 loads, resident for whole block) ----
    stage_w(lds + OFF_VW,  Vw,  kD*kD,   tid);
    stage_w(lds + OFF_UW,  Uw,  kD*kD,   tid);
    stage_w(lds + OFF_A1W, a1w, kD*2*kD, tid);
    stage_w(lds + OFF_A2W, a2w, 3*kD*kD, tid);

    // per-wave output column and biases (uniform per lane)
    const int col = wave * 16 + laneN;
    const float bVb    = Vb[col];
    const float bUb    = Ub[col];
    const float ba1    = a1b[col];
    const float ba2_vv = a2b[col];
    const float ba2_sv = a2b[kD + col];
    const float ba2_ss = a2b[2*kD + col];

    const long long outSBase = (long long)nTotal * 3 * kD;

    int buf = 0;
    for (int t = 0; t < kTilesPB; ++t) {
        const long long S0 = base0 + (long long)t * kTile;
        if (S0 >= nTotal) break;

        if (wave == 0) {
            __builtin_amdgcn_s_wait_tensorcnt(0);   // current buffer filled
            if (t + 1 < kTilesPB)                   // prefetch next tile into other buffer
                prefetch_tile(v, s, S0 + kTile, nTotal,
                              STAGE_BYTE + (buf ^ 1) * STAGE_FLOATS * 4);
        }
        __syncthreads();  // TDM data visible to all waves; prior-tile readers done

        // ---- in-place f32 -> bf16 conversion of staged tile ----
        float*  stf = (float*)((char*)lds + STAGE_BYTE + buf * STAGE_FLOATS * 4);
        bf16_t* Xbf = (bf16_t*)stf;          // bf16 X operand reuses low half
        v4f rg[8];
        const int ebase = tid * 32;          // 8192 floats / 256 threads
        #pragma unroll
        for (int j = 0; j < 8; ++j) rg[j] = *(const v4f*)(stf + ebase + j * 4);
        __syncthreads();                     // all reads done before overwrite
        if (ebase < 48 * kD) {               // v part -> Xbf (identity layout)
            #pragma unroll
            for (int j = 0; j < 8; ++j) {
                v4bf o;
                o[0]=(bf16_t)rg[j][0]; o[1]=(bf16_t)rg[j][1];
                o[2]=(bf16_t)rg[j][2]; o[3]=(bf16_t)rg[j][3];
                *(v4bf*)(Xbf + ebase + j * 4) = o;
            }
        } else {                             // s part -> s_stack columns 128..255
            int e = ebase - 48 * kD;
            int m = e >> 7, d = e & (kD - 1);
            bf16_t* dst = lds + OFF_SS + m * (2*kD) + kD + d;
            #pragma unroll
            for (int j = 0; j < 8; ++j) {
                v4bf o;
                o[0]=(bf16_t)rg[j][0]; o[1]=(bf16_t)rg[j][1];
                o[2]=(bf16_t)rg[j][2]; o[3]=(bf16_t)rg[j][3];
                *(v4bf*)(dst + j * 4) = o;
            }
        }
        __syncthreads();

        // ---- GEMM1: v1 = X @ Vw^T + Vb ----
        v8f v1acc[3] = {v8f{}, v8f{}, v8f{}};
        for (int k0 = 0; k0 < kD; k0 += 32) {
            v16bf b = load_b_frag(lds + OFF_VW, col, kD, k0, laneHi);
            for (int c = 0; c < 3; ++c) {
                v16bf a = load_a_frag(Xbf, c*16 + laneN, kD, k0, laneHi);
                v1acc[c] = WMMA_BF16(a, b, v1acc[c]);
            }
        }
        for (int c = 0; c < 3; ++c)
            for (int r = 0; r < 8; ++r) v1acc[c][r] += bVb;

        // spill v1 (bf16) so all waves see all K for GEMM2
        for (int c = 0; c < 3; ++c)
            for (int r = 0; r < 8; ++r)
                lds[OFF_V1 + (c*16 + r + 8*laneHi)*kD + col] = (bf16_t)v1acc[c][r];

        // vnorm over channels -> s_stack columns 0..127
        for (int r = 0; r < 8; ++r) {
            float nsq = 0.f;
            for (int c = 0; c < 3; ++c) nsq += v1acc[c][r] * v1acc[c][r];
            lds[OFF_SS + (r + 8*laneHi)*(2*kD) + col] = (bf16_t)sqrtf(nsq);
        }
        __syncthreads();

        // ---- GEMM2: u = v1 @ Uw^T + Ub ----
        v8f uacc[3] = {v8f{}, v8f{}, v8f{}};
        for (int k0 = 0; k0 < kD; k0 += 32) {
            v16bf b = load_b_frag(lds + OFF_UW, col, kD, k0, laneHi);
            for (int c = 0; c < 3; ++c) {
                v16bf a = load_a_frag(lds + OFF_V1, c*16 + laneN, kD, k0, laneHi);
                uacc[c] = WMMA_BF16(a, b, uacc[c]);
            }
        }
        for (int c = 0; c < 3; ++c)
            for (int r = 0; r < 8; ++r) uacc[c][r] += bUb;

        // ---- GEMM3: h = silu(s_stack @ a1w^T + a1b), K = 256 ----
        v8f hacc = v8f{};
        for (int k0 = 0; k0 < 2*kD; k0 += 32) {
            v16bf a = load_a_frag(lds + OFF_SS, laneN, 2*kD, k0, laneHi);
            v16bf b = load_b_frag(lds + OFF_A1W, col, 2*kD, k0, laneHi);
            hacc = WMMA_BF16(a, b, hacc);
        }
        for (int r = 0; r < 8; ++r) {
            float x = hacc[r] + ba1;
            x = x / (1.f + expf(-x));          // SiLU
            lds[OFF_H + (r + 8*laneHi)*kD + col] = (bf16_t)x;
        }
        __syncthreads();

        // ---- GEMM4: split = h @ a2w^T + a2b; wave w takes columns n, 128+n, 256+n ----
        v8f sp[3] = {v8f{}, v8f{}, v8f{}};
        for (int k0 = 0; k0 < kD; k0 += 32) {
            v16bf a = load_a_frag(lds + OFF_H, laneN, kD, k0, laneHi);
            for (int j = 0; j < 3; ++j) {
                v16bf b = load_b_frag(lds + OFF_A2W, j*kD + col, kD, k0, laneHi);
                sp[j] = WMMA_BF16(a, b, sp[j]);
            }
        }
        for (int r = 0; r < 8; ++r) {
            sp[0][r] += ba2_vv; sp[1][r] += ba2_sv; sp[2][r] += ba2_ss;
        }

        // ---- elementwise epilogue + coalesced-per-halfwave global stores ----
        for (int r = 0; r < 8; ++r) {
            long long smp = S0 + r + 8*laneHi;
            if (smp >= nTotal) continue;
            float avv = sp[0][r], asv = sp[1][r], ass = sp[2][r];
            float sdot = 0.f;
            for (int c = 0; c < 3; ++c) {
                float uv = uacc[c][r];
                out[(smp*3 + c)*kD + col] = uv * avv;        // out_v
                sdot += uv * v1acc[c][r];
            }
            out[outSBase + smp*kD + col] = sdot * asv + ass; // out_s
        }
        buf ^= 1;
    }
}

extern "C" void kernel_launch(void* const* d_in, const int* in_sizes, int n_in,
                              void* d_out, int out_size, void* d_ws, size_t ws_size,
                              hipStream_t stream) {
    const float* s   = (const float*)d_in[0];
    const float* v   = (const float*)d_in[1];
    const float* Vw  = (const float*)d_in[2];
    const float* Vb  = (const float*)d_in[3];
    const float* Uw  = (const float*)d_in[4];
    const float* Ub  = (const float*)d_in[5];
    const float* a1w = (const float*)d_in[6];
    const float* a1b = (const float*)d_in[7];
    const float* a2w = (const float*)d_in[8];
    const float* a2b = (const float*)d_in[9];
    float* out = (float*)d_out;

    const int nTotal = in_sizes[0] / kD;   // s is [N,1,D]
    const int grid = (nTotal + kSamplesPB - 1) / kSamplesPB;

    (void)hipFuncSetAttribute((const void*)painn_fused_kernel,
                              hipFuncAttributeMaxDynamicSharedMemorySize, LDS_BYTES);
    painn_fused_kernel<<<grid, kThreads, LDS_BYTES, stream>>>(
        s, v, Vw, Vb, Uw, Ub, a1w, a1b, a2w, a2b, out, nTotal);
}
